// MeshODENet_64819646431388
// MI455X (gfx1250) — matrix operand order repaired and checked by reading, not compiled, not run on hardware
//
#include <hip/hip_runtime.h>

#define DEVINL __device__ __forceinline__

typedef __attribute__((ext_vector_type(16))) _Float16 v16h;
typedef __attribute__((ext_vector_type(8)))  _Float16 v8h;
typedef __attribute__((ext_vector_type(8)))  float    v8f;

// ---------------------------------------------------------------------------
// WMMA helpers (v_wmma_f32_16x16x32_f16), wave32.
//
// A fragment (16x32 f16, row-major tile in LDS, stride in halfs):
//   lane L: m=L&15, hsel=L>>4; elem j : k = k0 + 16*hsel + j   (contig 32B)
// B fragment (32x16 f16) from TRANSPOSED weights Wt[N][K] (row-major, f16):
//   lane L: n=L&15, hsel=L>>4; elem j = W[k0 + 16*hsel + j][n0+n]
//          = Wt[(n0+n)*K + k0 + 16*hsel + j]   (contig 32B)
// C/D (16x16 f32): lane L: col n = L&15 (+n0), elem r: row m = 8*(L>>4)+r
// ---------------------------------------------------------------------------
DEVINL v16h load_a_frag(const _Float16* row, int k0, int hsel) {
    union { v16h v; v8h h[2]; } u;
    u.h[0] = *(const v8h*)(row + k0 + 16 * hsel);
    u.h[1] = *(const v8h*)(row + k0 + 16 * hsel + 8);
    return u.v;
}

DEVINL v8f gemm_tile(const _Float16* X, int xstride, int K,
                     const _Float16* Wt, int n0, int lane) {
    int m = lane & 15, hsel = lane >> 4;
    v8f acc = {};
    const _Float16* xrow = X + m * xstride;
    const _Float16* wrow = Wt + (size_t)(n0 + m) * K + 16 * hsel;
#pragma unroll
    for (int k0 = 0; k0 < K; k0 += 32) {
        v16h a = load_a_frag(xrow, k0, hsel);
        v16h b = *(const v16h*)(wrow + k0);
        acc = __builtin_amdgcn_wmma_f32_16x16x32_f16(
            false, a, false, b, (short)0, acc, false, false);
    }
    return acc;
}

DEVINL void atomic_add_f32(float* p, float v) {
    __hip_atomic_fetch_add(p, v, __ATOMIC_RELAXED, __HIP_MEMORY_SCOPE_AGENT);
}

// gfx1250 async gather: global -> LDS, 16B per lane, tracked by ASYNCcnt.
// VDST operand holds the LDS byte offset (HW adds the wave LDS base); the
// low 32 bits of a generic pointer to __shared__ are exactly that offset.
DEVINL void async_ld_b128(void* lds_dst, const void* src) {
    unsigned int off = (unsigned int)(uintptr_t)lds_dst;
    asm volatile("global_load_async_to_lds_b128 %0, %1, off"
                 :: "v"(off), "v"(src) : "memory");
}
DEVINL void wait_async0() {
    asm volatile("s_wait_asynccnt 0" ::: "memory");
}

// ---------------------------------------------------------------------------
// one-time: f32 [K][64] weight -> f16 transposed [64][K]
// ---------------------------------------------------------------------------
__global__ void k_wt(const float* __restrict__ w, _Float16* __restrict__ wt, int K) {
    int i = blockIdx.x * blockDim.x + threadIdx.x;
    if (i >= K * 64) return;
    int k = i >> 6, n = i & 63;
    wt[(size_t)n * K + k] = (_Float16)w[(size_t)k * 64 + n];
}

// one-time: static part of node-encoder layer1 (youngs + one-hot columns + bias)
__global__ void k_h1pre(const float* __restrict__ youngs, const float* ym, const float* ys,
                        const float* __restrict__ onehot, const float* __restrict__ w_ne1,
                        const float* __restrict__ b_ne1, float* __restrict__ h1pre,
                        int N, int Npad) {
    int idx = blockIdx.x * blockDim.x + threadIdx.x;
    if (idx >= Npad * 64) return;
    int node = idx >> 6, o = idx & 63;
    int nc = node < N ? node : N - 1;
    float s = b_ne1[o];
    s += ((youngs[nc] - ym[0]) / ys[0]) * w_ne1[3 * 64 + o];
#pragma unroll
    for (int i = 0; i < 9; ++i) s += onehot[nc * 9 + i] * w_ne1[(4 + i) * 64 + o];
    h1pre[idx] = s;
}

// ---------------------------------------------------------------------------
// one-time: edge encoder  e0 = relu(eattr_n @ w_ee1 + b1) @ w_ee2 + b2
// ---------------------------------------------------------------------------
__global__ void k_edge_encode(const float* __restrict__ eattr, const float* mm, const float* ms,
                              const float* __restrict__ w_ee1, const float* __restrict__ b_ee1,
                              const _Float16* __restrict__ wt_ee2, const float* __restrict__ b_ee2,
                              float* __restrict__ e0f, _Float16* __restrict__ e0h, int E) {
    __shared__ __align__(16) _Float16 Xt[64][64];
    int tid = threadIdx.x;
    int eb = blockIdx.x * 64;
    bool full = (eb + 64 <= E);
    {
        int el = tid >> 2, og = (tid & 3) * 16;
        int e = eb + el; if (e >= E) e = E - 1;
        float en[4];
#pragma unroll
        for (int i = 0; i < 4; ++i) en[i] = (eattr[(size_t)e * 4 + i] - mm[i]) / ms[i];
#pragma unroll
        for (int r = 0; r < 16; ++r) {
            int o = og + r;
            float s = b_ee1[o];
#pragma unroll
            for (int i = 0; i < 4; ++i) s += en[i] * w_ee1[i * 64 + o];
            Xt[el][o] = (_Float16)(s > 0.f ? s : 0.f);
        }
    }
    __syncthreads();
    int lane = tid & 31, wave = tid >> 5;
#pragma unroll
    for (int t = wave; t < 16; t += 8) {
        int rt = t >> 2, ct = t & 3;
        v8f acc = gemm_tile(&Xt[rt * 16][0], 64, 64, wt_ee2, ct * 16, lane);
        int n = ct * 16 + (lane & 15), mb = (lane >> 4) * 8;
        float bn = b_ee2[n];
        if (full) {
#pragma unroll
            for (int r = 0; r < 8; ++r) {
                size_t e = (size_t)(eb + rt * 16 + mb + r);
                float v = acc[r] + bn;
                e0f[e * 64 + n] = v;
                e0h[e * 64 + n] = (_Float16)v;
            }
        } else {
#pragma unroll
            for (int r = 0; r < 8; ++r) {
                int e = eb + rt * 16 + mb + r;
                if (e < E) {
                    float v = acc[r] + bn;
                    e0f[(size_t)e * 64 + n] = v;
                    e0h[(size_t)e * 64 + n] = (_Float16)v;
                }
            }
        }
    }
}

// ---------------------------------------------------------------------------
// per-deriv: node encoder (with fused RK4 axpy on the input velocity)
//   veval = vbase + coef*dt*kprev ;  vn = (veval - vmean)/vstd
//   h = relu(h1pre + vn @ w_ne1[0:3]) @ w_ne2 + b2
// ---------------------------------------------------------------------------
__global__ void k_node_encode(const float* __restrict__ vbase, const float* __restrict__ kprev,
                              const float* __restrict__ ts, int step, float coef,
                              const float* vm, const float* vs,
                              const float* __restrict__ w_ne1, const float* __restrict__ h1pre,
                              const _Float16* __restrict__ wt_ne2, const float* __restrict__ b_ne2,
                              float* __restrict__ hf, _Float16* __restrict__ hh, int N) {
    __shared__ __align__(16) _Float16 Xt[64][64];
    int tid = threadIdx.x;
    int nb = blockIdx.x * 64;
    {
        float dt = ts[step + 1] - ts[step];
        int nl = tid >> 2, og = (tid & 3) * 16;
        int node = nb + nl; if (node >= N) node = N - 1;
        float vn[3];
#pragma unroll
        for (int i = 0; i < 3; ++i) {
            float vv = vbase[(size_t)node * 3 + i] + coef * dt * kprev[(size_t)node * 3 + i];
            vn[i] = (vv - vm[i]) / vs[i];
        }
#pragma unroll
        for (int r = 0; r < 16; ++r) {
            int o = og + r;
            float s = h1pre[(size_t)node * 64 + o];
#pragma unroll
            for (int i = 0; i < 3; ++i) s += vn[i] * w_ne1[i * 64 + o];
            Xt[nl][o] = (_Float16)(s > 0.f ? s : 0.f);
        }
    }
    __syncthreads();
    int lane = tid & 31, wave = tid >> 5;
#pragma unroll
    for (int t = wave; t < 16; t += 8) {
        int rt = t >> 2, ct = t & 3;
        v8f acc = gemm_tile(&Xt[rt * 16][0], 64, 64, wt_ne2, ct * 16, lane);
        int n = ct * 16 + (lane & 15), mb = (lane >> 4) * 8;
        float bn = b_ne2[n];
#pragma unroll
        for (int r = 0; r < 8; ++r) {
            size_t row = (size_t)(nb + rt * 16 + mb + r);   // padded buffers: in range
            float v = acc[r] + bn;
            hf[row * 64 + n] = v;
            hh[row * 64 + n] = (_Float16)v;
        }
    }
}

// ---------------------------------------------------------------------------
// per-deriv: edge message + scatter
//   msg = relu([e0,h_src,h_dst] @ w_em1 + b1) @ w_em2 + b2
//   agg[dst] += e0 + msg      (fp32 global atomics)
// Gather tile is filled with gfx1250 async load-to-LDS (ASYNCcnt).
// ---------------------------------------------------------------------------
__global__ void k_edge_message(const int* __restrict__ ei,
                               const _Float16* __restrict__ e0h, const float* __restrict__ e0f,
                               const _Float16* __restrict__ hh,
                               const _Float16* __restrict__ wt_em1, const float* __restrict__ b_em1,
                               const _Float16* __restrict__ wt_em2, const float* __restrict__ b_em2,
                               float* __restrict__ agg, int E) {
    __shared__ __align__(16) _Float16 Xt[64][192];
    __shared__ __align__(16) _Float16 Ht[64][64];
    __shared__ int srcl[64];
    __shared__ int dstl[64];
    int tid = threadIdx.x;
    int eb = blockIdx.x * 64;
    bool full = (eb + 64 <= E);
    if (tid < 64) {
        int e = eb + tid; if (e >= E) e = E - 1;
        srcl[tid] = ei[e];
        dstl[tid] = ei[E + e];
    }
    __syncthreads();
    // build X = [e0 | h_src | h_dst]: 24 async 16B gathers per edge, direct to LDS
#pragma unroll
    for (int it = 0; it < 6; ++it) {
        int idx = tid + it * 256;                  // 64*24 == 1536 == 6*256
        int el = idx / 24, c = idx % 24;
        int e = eb + el; if (e >= E) e = E - 1;
        const _Float16* sp;
        if (c < 8)       sp = e0h + (size_t)e * 64 + c * 8;
        else if (c < 16) sp = hh + (size_t)srcl[el] * 64 + (c - 8) * 8;
        else             sp = hh + (size_t)dstl[el] * 64 + (c - 16) * 8;
        async_ld_b128(&Xt[el][c * 8], sp);
    }
    wait_async0();
    __syncthreads();
    int lane = tid & 31, wave = tid >> 5;
#pragma unroll
    for (int t = wave; t < 16; t += 8) {           // layer 1, K=192
        int rt = t >> 2, ct = t & 3;
        v8f acc = gemm_tile(&Xt[rt * 16][0], 192, 192, wt_em1, ct * 16, lane);
        int n = ct * 16 + (lane & 15), mb = (lane >> 4) * 8;
        float bn = b_em1[n];
#pragma unroll
        for (int r = 0; r < 8; ++r) {
            float v = acc[r] + bn;
            Ht[rt * 16 + mb + r][n] = (_Float16)(v > 0.f ? v : 0.f);
        }
    }
    __syncthreads();
#pragma unroll
    for (int t = wave; t < 16; t += 8) {           // layer 2, K=64, + residual, scatter
        int rt = t >> 2, ct = t & 3;
        v8f acc = gemm_tile(&Ht[rt * 16][0], 64, 64, wt_em2, ct * 16, lane);
        int n = ct * 16 + (lane & 15), mb = (lane >> 4) * 8;
        float bn = b_em2[n];
        if (full) {
#pragma unroll
            for (int r = 0; r < 8; ++r) {
                int el = rt * 16 + mb + r;
                size_t e = (size_t)(eb + el);
                float v = e0f[e * 64 + n] + acc[r] + bn;
                atomic_add_f32(&agg[(size_t)dstl[el] * 64 + n], v);
            }
        } else {
#pragma unroll
            for (int r = 0; r < 8; ++r) {
                int el = rt * 16 + mb + r;
                int e = eb + el;
                if (e < E) {
                    float v = e0f[(size_t)e * 64 + n] + acc[r] + bn;
                    atomic_add_f32(&agg[(size_t)dstl[el] * 64 + n], v);
                }
            }
        }
    }
}

// ---------------------------------------------------------------------------
// per-deriv: node update + decoder
//   h2 = h + relu([h,agg] @ w_nm1 + b1) @ w_nm2 + b2
//   acc = (relu(h2 @ w_dec1 + b1) @ w_dec2 + b2) * acc_std + acc_mean
// ---------------------------------------------------------------------------
__global__ void k_node_update(const _Float16* __restrict__ hh, const float* __restrict__ hf,
                              const float* __restrict__ agg,
                              const _Float16* __restrict__ wt_nm1, const float* __restrict__ b_nm1,
                              const _Float16* __restrict__ wt_nm2, const float* __restrict__ b_nm2,
                              const _Float16* __restrict__ wt_dc1, const float* __restrict__ b_dec1,
                              const float* __restrict__ w_dec2, const float* __restrict__ b_dec2,
                              const float* am, const float* as,
                              float* __restrict__ kout, int N) {
    __shared__ __align__(16) _Float16 Xt[64][128];
    __shared__ __align__(16) _Float16 Ht[64][64];
    __shared__ __align__(16) _Float16 H2[64][64];
    int tid = threadIdx.x;
    int nb = blockIdx.x * 64;
    // left half of X: h rows, async 16B copies straight to LDS
#pragma unroll
    for (int it = 0; it < 2; ++it) {
        int idx = tid + it * 256;                  // 64*8 == 512
        int nl = idx >> 3, c = idx & 7;
        async_ld_b128(&Xt[nl][c * 8], hh + (size_t)(nb + nl) * 64 + c * 8);
    }
    // right half: agg f32 -> f16 (needs VALU convert)
#pragma unroll
    for (int it = 0; it < 16; ++it) {
        int idx = tid + it * 256;                  // 64*64 == 4096
        int nl = idx >> 6, c = idx & 63;
        Xt[nl][64 + c] = (_Float16)agg[(size_t)(nb + nl) * 64 + c];
    }
    wait_async0();
    __syncthreads();
    int lane = tid & 31, wave = tid >> 5;
#pragma unroll
    for (int t = wave; t < 16; t += 8) {           // nm layer1, K=128
        int rt = t >> 2, ct = t & 3;
        v8f acc = gemm_tile(&Xt[rt * 16][0], 128, 128, wt_nm1, ct * 16, lane);
        int n = ct * 16 + (lane & 15), mb = (lane >> 4) * 8;
        float bn = b_nm1[n];
#pragma unroll
        for (int r = 0; r < 8; ++r) {
            float v = acc[r] + bn;
            Ht[rt * 16 + mb + r][n] = (_Float16)(v > 0.f ? v : 0.f);
        }
    }
    __syncthreads();
#pragma unroll
    for (int t = wave; t < 16; t += 8) {           // nm layer2 + residual
        int rt = t >> 2, ct = t & 3;
        v8f acc = gemm_tile(&Ht[rt * 16][0], 64, 64, wt_nm2, ct * 16, lane);
        int n = ct * 16 + (lane & 15), mb = (lane >> 4) * 8;
        float bn = b_nm2[n];
#pragma unroll
        for (int r = 0; r < 8; ++r) {
            int nl = rt * 16 + mb + r;
            float v = hf[(size_t)(nb + nl) * 64 + n] + acc[r] + bn;
            H2[nl][n] = (_Float16)v;
        }
    }
    __syncthreads();
    _Float16* Dd = &Xt[0][0];                      // reuse Xt as [64][64] scratch
#pragma unroll
    for (int t = wave; t < 16; t += 8) {           // dec layer1
        int rt = t >> 2, ct = t & 3;
        v8f acc = gemm_tile(&H2[rt * 16][0], 64, 64, wt_dc1, ct * 16, lane);
        int n = ct * 16 + (lane & 15), mb = (lane >> 4) * 8;
        float bn = b_dec1[n];
#pragma unroll
        for (int r = 0; r < 8; ++r) {
            float v = acc[r] + bn;
            Dd[(rt * 16 + mb + r) * 64 + n] = (_Float16)(v > 0.f ? v : 0.f);
        }
    }
    __syncthreads();
    if (tid < 192) {                               // dec layer2 (64->3), VALU
        int nl = tid / 3, j = tid % 3;
        int node = nb + nl;
        if (node < N) {
            float s = b_dec2[j];
#pragma unroll
            for (int c = 0; c < 64; ++c) s += (float)Dd[nl * 64 + c] * w_dec2[c * 3 + j];
            kout[(size_t)node * 3 + j] = s * as[j] + am[j];
        }
    }
}

// ---------------------------------------------------------------------------
// RK4 glue
// ---------------------------------------------------------------------------
__global__ void k_init(const float* __restrict__ wp, const float* __restrict__ v0,
                       float* pos, float* vel, float* opos, float* ovel, int n3) {
    int i = blockIdx.x * blockDim.x + threadIdx.x;
    if (i >= n3) return;
    float p = wp[i], v = v0[i];
    pos[i] = p; opos[i] = p; vel[i] = v; ovel[i] = v;
}

// pos += dt*vel + dt^2/6*(k1+k2+k3) ; vel += dt/6*(k1+2k2+2k3+k4)
__global__ void k_combine(const float* __restrict__ ts, int step,
                          const float* __restrict__ k1, const float* __restrict__ k2,
                          const float* __restrict__ k3, const float* __restrict__ k4,
                          float* pos, float* vel, float* opos, float* ovel, int n3) {
    int i = blockIdx.x * blockDim.x + threadIdx.x;
    if (i >= n3) return;
    float dt = ts[step + 1] - ts[step];
    float p = pos[i] + dt * vel[i] + (dt * dt / 6.f) * (k1[i] + k2[i] + k3[i]);
    float v = vel[i] + (dt / 6.f) * (k1[i] + 2.f * k2[i] + 2.f * k3[i] + k4[i]);
    pos[i] = p; vel[i] = v; opos[i] = p; ovel[i] = v;
}

// ---------------------------------------------------------------------------
extern "C" void kernel_launch(void* const* d_in, const int* in_sizes, int n_in,
                              void* d_out, int out_size, void* d_ws, size_t ws_size,
                              hipStream_t stream) {
    (void)n_in; (void)out_size; (void)ws_size;
    const float* world_pos = (const float*)d_in[0];
    const float* velocity  = (const float*)d_in[1];
    const int*   edge_index= (const int*)d_in[2];
    const float* eattr     = (const float*)d_in[3];
    const float* youngs    = (const float*)d_in[4];
    const float* one_hot   = (const float*)d_in[5];
    const float* t_span    = (const float*)d_in[7];
    const float* ymean = (const float*)d_in[8],  *ystd = (const float*)d_in[9];
    const float* memean= (const float*)d_in[10], *mestd= (const float*)d_in[11];
    const float* vmean = (const float*)d_in[12], *vstd = (const float*)d_in[13];
    const float* amean = (const float*)d_in[14], *astd = (const float*)d_in[15];
    const float* w_ne1 = (const float*)d_in[16], *b_ne1 = (const float*)d_in[17];
    const float* w_ne2 = (const float*)d_in[18], *b_ne2 = (const float*)d_in[19];
    const float* w_ee1 = (const float*)d_in[20], *b_ee1 = (const float*)d_in[21];
    const float* w_ee2 = (const float*)d_in[22], *b_ee2 = (const float*)d_in[23];
    const float* w_em1 = (const float*)d_in[24], *b_em1 = (const float*)d_in[25];
    const float* w_em2 = (const float*)d_in[26], *b_em2 = (const float*)d_in[27];
    const float* w_nm1 = (const float*)d_in[28], *b_nm1 = (const float*)d_in[29];
    const float* w_nm2 = (const float*)d_in[30], *b_nm2 = (const float*)d_in[31];
    const float* w_dc1 = (const float*)d_in[32], *b_dc1 = (const float*)d_in[33];
    const float* w_dc2 = (const float*)d_in[34], *b_dc2 = (const float*)d_in[35];

    const int N = in_sizes[0] / 3;
    const int E = in_sizes[2] / 2;
    const int T = in_sizes[7];
    const int nbN = (N + 63) / 64, Npad = nbN * 64;
    const int nbE = (E + 63) / 64;
    const int n3 = N * 3;

    char* ws = (char*)d_ws;
    size_t off = 0;
    auto alloc = [&](size_t bytes) -> char* {
        char* p = ws + off;
        off = (off + bytes + 255) & ~(size_t)255;
        return p;
    };
    _Float16* wtNE2 = (_Float16*)alloc((size_t)64 * 64 * 2);
    _Float16* wtEE2 = (_Float16*)alloc((size_t)64 * 64 * 2);
    _Float16* wtEM1 = (_Float16*)alloc((size_t)192 * 64 * 2);
    _Float16* wtEM2 = (_Float16*)alloc((size_t)64 * 64 * 2);
    _Float16* wtNM1 = (_Float16*)alloc((size_t)128 * 64 * 2);
    _Float16* wtNM2 = (_Float16*)alloc((size_t)64 * 64 * 2);
    _Float16* wtDC1 = (_Float16*)alloc((size_t)64 * 64 * 2);
    float*    h1pre = (float*)alloc((size_t)Npad * 64 * 4);
    float*    e0f   = (float*)alloc((size_t)E * 64 * 4);
    _Float16* e0h   = (_Float16*)alloc((size_t)E * 64 * 2);
    float*    hf    = (float*)alloc((size_t)Npad * 64 * 4);
    _Float16* hh    = (_Float16*)alloc((size_t)Npad * 64 * 2);
    float*    agg   = (float*)alloc((size_t)Npad * 64 * 4);
    float*    pos   = (float*)alloc((size_t)n3 * 4);
    float*    vel   = (float*)alloc((size_t)n3 * 4);
    float*    k1    = (float*)alloc((size_t)n3 * 4);
    float*    k2    = (float*)alloc((size_t)n3 * 4);
    float*    k3    = (float*)alloc((size_t)n3 * 4);
    float*    k4    = (float*)alloc((size_t)n3 * 4);

    // one-time prep (cheap; re-runs each graph replay)
    k_wt<<<(64 * 64 + 255) / 256, 256, 0, stream>>>(w_ne2, wtNE2, 64);
    k_wt<<<(64 * 64 + 255) / 256, 256, 0, stream>>>(w_ee2, wtEE2, 64);
    k_wt<<<(192 * 64 + 255) / 256, 256, 0, stream>>>(w_em1, wtEM1, 192);
    k_wt<<<(64 * 64 + 255) / 256, 256, 0, stream>>>(w_em2, wtEM2, 64);
    k_wt<<<(128 * 64 + 255) / 256, 256, 0, stream>>>(w_nm1, wtNM1, 128);
    k_wt<<<(64 * 64 + 255) / 256, 256, 0, stream>>>(w_nm2, wtNM2, 64);
    k_wt<<<(64 * 64 + 255) / 256, 256, 0, stream>>>(w_dc1, wtDC1, 64);
    k_h1pre<<<(Npad * 64 + 255) / 256, 256, 0, stream>>>(youngs, ymean, ystd, one_hot,
                                                         w_ne1, b_ne1, h1pre, N, Npad);
    k_edge_encode<<<nbE, 256, 0, stream>>>(eattr, memean, mestd, w_ee1, b_ee1,
                                           wtEE2, b_ee2, e0f, e0h, E);

    float* outp = (float*)d_out;
    float* outv = outp + (size_t)T * n3;
    k_init<<<(n3 + 255) / 256, 256, 0, stream>>>(world_pos, velocity, pos, vel, outp, outv, n3);

    for (int step = 0; step < T - 1; ++step) {
        auto deriv = [&](const float* kprev, float coef, float* kout) {
            k_node_encode<<<nbN, 256, 0, stream>>>(vel, kprev, t_span, step, coef,
                                                   vmean, vstd, w_ne1, h1pre,
                                                   wtNE2, b_ne2, hf, hh, N);
            hipMemsetAsync(agg, 0, (size_t)Npad * 64 * 4, stream);
            k_edge_message<<<nbE, 256, 0, stream>>>(edge_index, e0h, e0f, hh,
                                                    wtEM1, b_em1, wtEM2, b_em2, agg, E);
            k_node_update<<<nbN, 256, 0, stream>>>(hh, hf, agg, wtNM1, b_nm1, wtNM2, b_nm2,
                                                   wtDC1, b_dc1, w_dc2, b_dc2,
                                                   amean, astd, kout, N);
        };
        deriv(vel, 0.0f, k1);
        deriv(k1, 0.5f, k2);
        deriv(k2, 0.5f, k3);
        deriv(k3, 1.0f, k4);
        k_combine<<<(n3 + 255) / 256, 256, 0, stream>>>(t_span, step, k1, k2, k3, k4, pos, vel,
                                                        outp + (size_t)(step + 1) * n3,
                                                        outv + (size_t)(step + 1) * n3, n3);
    }
}
